// NCALoss_72567767433948
// MI455X (gfx1250) — compile-verified, matrix operand order checked by reading, not run on hardware
//
#include <hip/hip_runtime.h>
#include <math.h>

// ---------------------------------------------------------------------------
// NCA loss on MI455X (gfx1250, wave32).
//   cross = feature @ proxy^T via V_WMMA_F32_16X16X32_F16 with split-precision
//   (hi+lo f16) 3-pass accumulation -> near-fp32 accuracy, memory-bound.
//   loss_row = d_pos + log(sum_c exp(-d_c) - exp(-d_pos))
// Deterministic 2-kernel reduction (block partials -> fixed-order tree).
// ---------------------------------------------------------------------------

#define BATCH 131072
#define CLS   45
#define CPAD  48
#define DIM   512
#define ROWS_PER_BLOCK 128
#define THREADS 256
#define EPSV 1e-6f

typedef _Float16 v8h  __attribute__((ext_vector_type(8)));
typedef _Float16 v16h __attribute__((ext_vector_type(16)));
typedef float    v8f  __attribute__((ext_vector_type(8)));

// dynamic LDS layout (bytes)
#define OFF_PHI 0
#define OFF_PLO (CPAD * DIM * 2)                 // 49152
#define OFF_P2  (OFF_PLO + CPAD * DIM * 2)       // 98304
#define OFF_SP  (OFF_P2 + CPAD * 4)
#define OFF_PP  (OFF_SP + CPAD * 4)
#define OFF_PS  (OFF_PP + 192 * 4)
#define OFF_F2  (OFF_PS + 192 * 4)
#define OFF_SF  (OFF_F2 + ROWS_PER_BLOCK * 4)
#define OFF_WS  (OFF_SF + ROWS_PER_BLOCK * 4)
#define SMEM_BYTES (OFF_WS + 8 * 4)              // 101280 B < 320KB WGP LDS

__device__ __forceinline__ void cvt8(const float4 a, const float4 b,
                                     v8h& hi, v8h& lo, float& s2, float& s1) {
  float v[8] = {a.x, a.y, a.z, a.w, b.x, b.y, b.z, b.w};
#pragma unroll
  for (int i = 0; i < 8; ++i) {
    float x = v[i];
    s2 += x * x;
    s1 += x;
    _Float16 h = (_Float16)x;
    hi[i] = h;
    lo[i] = (_Float16)(x - (float)h);   // exact residual (hi within 2x of x)
  }
}

#define CAT16(lo8, hi8) \
  __builtin_shufflevector(lo8, hi8, 0, 1, 2, 3, 4, 5, 6, 7, 8, 9, 10, 11, 12, 13, 14, 15)

__global__ __launch_bounds__(THREADS)
void nca_main(const float* __restrict__ feature, const int* __restrict__ label,
              const float* __restrict__ proxy, float* __restrict__ partial) {
  extern __shared__ char smem[];
  _Float16* Phi = (_Float16*)(smem + OFF_PHI);
  _Float16* Plo = (_Float16*)(smem + OFF_PLO);
  float* p2s  = (float*)(smem + OFF_P2);
  float* sps  = (float*)(smem + OFF_SP);
  float* pp   = (float*)(smem + OFF_PP);
  float* ps   = (float*)(smem + OFF_PS);
  float* f2b  = (float*)(smem + OFF_F2);
  float* sfb  = (float*)(smem + OFF_SF);
  float* wsum = (float*)(smem + OFF_WS);

  const int tid = threadIdx.x;

  // --- stage proxy into LDS as f16 hi/lo split (classes 45..47 = 0) -------
  for (int i = tid; i < CPAD * DIM; i += THREADS) {
    int c = i >> 9;
    float x = (c < CLS) ? proxy[i] : 0.0f;
    _Float16 h = (_Float16)x;
    Phi[i] = h;
    Plo[i] = (_Float16)(x - (float)h);
  }

  // --- p2[c], sp[c]: 4 threads per class, then combine --------------------
  if (tid < 192) {
    int c = tid >> 2, q = tid & 3;
    float a2 = 0.f, a1 = 0.f;
    if (c < CLS) {
      const float* pr = proxy + c * DIM + q * 128;
      for (int k = 0; k < 128; ++k) { float x = pr[k]; a2 += x * x; a1 += x; }
    }
    pp[tid] = a2;
    ps[tid] = a1;
  }
  __syncthreads();
  if (tid < CPAD) {
    int b = tid * 4;
    p2s[tid] = pp[b] + pp[b + 1] + pp[b + 2] + pp[b + 3];
    sps[tid] = ps[b] + ps[b + 1] + ps[b + 2] + ps[b + 3];
  }
  __syncthreads();

  // --- GEMM: each wave owns a 16-row M tile, 3 N tiles, K=512 -------------
  const int wave = tid >> 5;
  const int lane = tid & 31;
  const int laneLow = lane & 15;
  const int laneHi = lane >> 4;           // 0: K 0..7/16..23 ; 1: K 8..15/24..31
  const int rowBase = blockIdx.x * ROWS_PER_BLOCK + wave * 16;
  const float* frow = feature + (size_t)(rowBase + laneLow) * DIM;

  v8f acc0 = {}, acc1 = {}, acc2 = {};
  float f2p = 0.f, sfp = 0.f;

  for (int kc = 0; kc < 16; ++kc) {
    const int base = kc * 32 + laneHi * 8;
    float4 x0 = *(const float4*)(frow + base);
    float4 x1 = *(const float4*)(frow + base + 4);
    float4 x2 = *(const float4*)(frow + base + 16);
    float4 x3 = *(const float4*)(frow + base + 20);

    v8h ah0 = {}, al0 = {}, ah1 = {}, al1 = {};
    cvt8(x0, x1, ah0, al0, f2p, sfp);
    cvt8(x2, x3, ah1, al1, f2p, sfp);
    v16h ahi = CAT16(ah0, ah1);
    v16h alo = CAT16(al0, al1);

#pragma unroll
    for (int t = 0; t < 3; ++t) {
      const int n = t * 16 + laneLow;
      const _Float16* bh = Phi + n * DIM + base;
      const _Float16* bl = Plo + n * DIM + base;
      v8h bh0 = *(const v8h*)bh;
      v8h bh1 = *(const v8h*)(bh + 16);
      v8h bl0 = *(const v8h*)bl;
      v8h bl1 = *(const v8h*)(bl + 16);
      v16h bhi = CAT16(bh0, bh1);
      v16h blo = CAT16(bl0, bl1);

      v8f& acc = (t == 0) ? acc0 : ((t == 1) ? acc1 : acc2);
      // smallest terms first: lo*hi + hi*lo + hi*hi
      acc = __builtin_amdgcn_wmma_f32_16x16x32_f16(false, alo, false, bhi,
                                                   (short)0, acc, false, false);
      acc = __builtin_amdgcn_wmma_f32_16x16x32_f16(false, ahi, false, blo,
                                                   (short)0, acc, false, false);
      acc = __builtin_amdgcn_wmma_f32_16x16x32_f16(false, ahi, false, bhi,
                                                   (short)0, acc, false, false);
    }
  }

  // combine the two K-halves of each row's stats across lane halves
  float f2 = f2p + __shfl_xor(f2p, 16, 32);
  float sf = sfp + __shfl_xor(sfp, 16, 32);
  if (laneHi == 0) {
    f2b[wave * 16 + laneLow] = f2;
    sfb[wave * 16 + laneLow] = sf;
  }
  // same-wave LDS producer/consumer: DS ops are in-order within a wave

  int labv[8];
  float f2r[8], sfr[8];
#pragma unroll
  for (int r = 0; r < 8; ++r) {
    int M = r + laneHi * 8;               // C/D layout: VGPR r -> M=r (+8 hi half)
    labv[r] = label[rowBase + M];
    f2r[r] = f2b[wave * 16 + M];
    sfr[r] = sfb[wave * 16 + M];
  }

  float sume[8], pose[8], dpos[8];
#pragma unroll
  for (int r = 0; r < 8; ++r) { sume[r] = 0.f; pose[r] = 0.f; dpos[r] = 0.f; }

  const float k0 = (float)DIM * EPSV * EPSV;
#pragma unroll
  for (int t = 0; t < 3; ++t) {
    int c = t * 16 + laneLow;
    v8f accv = (t == 0) ? acc0 : ((t == 1) ? acc1 : acc2);
    if (c < CLS) {
      float p2c = p2s[c], spc = sps[c];
#pragma unroll
      for (int r = 0; r < 8; ++r) {
        float d2 = f2r[r] + p2c - 2.f * accv[r] +
                   2.f * EPSV * (sfr[r] - spc) + k0;
        float d = sqrtf(fmaxf(d2, 0.f));
        float e = expf(-d);
        sume[r] += e;
        if (labv[r] == c) { pose[r] += e; dpos[r] += d; }
      }
    }
  }

  // reduce classes across the 16-lane N group; pick one lane per row
  float local = 0.f;
#pragma unroll
  for (int r = 0; r < 8; ++r) {
    float se = sume[r], pe = pose[r], dp = dpos[r];
#pragma unroll
    for (int m = 1; m < 16; m <<= 1) {
      se += __shfl_xor(se, m, 32);
      pe += __shfl_xor(pe, m, 32);
      dp += __shfl_xor(dp, m, 32);
    }
    float lossr = dp + logf(se - pe);     // -log(pos/den), pos = exp(-d_pos)
    if (laneLow == r) local += lossr;     // lanes r and r+16 own rows r, r+8
  }
#pragma unroll
  for (int m = 1; m < 32; m <<= 1) local += __shfl_xor(local, m, 32);
  if (lane == 0) wsum[wave] = local;
  __syncthreads();
  if (tid == 0) {
    float s = 0.f;
#pragma unroll
    for (int w = 0; w < 8; ++w) s += wsum[w];   // fixed order -> deterministic
    partial[blockIdx.x] = s;
  }
}

__global__ __launch_bounds__(256)
void nca_reduce(const float* __restrict__ partial, float* __restrict__ out) {
  __shared__ float red[256];
  int t = threadIdx.x;
  // 1024 block partials, fixed-order tree -> deterministic
  float s = partial[t] + partial[t + 256] + partial[t + 512] + partial[t + 768];
  red[t] = s;
  __syncthreads();
  for (int off = 128; off > 0; off >>= 1) {
    if (t < off) red[t] += red[t + off];
    __syncthreads();
  }
  if (t == 0) out[0] = red[0];
}

extern "C" void kernel_launch(void* const* d_in, const int* in_sizes, int n_in,
                              void* d_out, int out_size, void* d_ws, size_t ws_size,
                              hipStream_t stream) {
  const float* feature = (const float*)d_in[0];
  const int*   label   = (const int*)d_in[1];   // jax default x64-off -> int32
  const float* proxy   = (const float*)d_in[2];
  float* out = (float*)d_out;
  float* partial = (float*)d_ws;                // 1024 floats = 4 KB scratch

  nca_main<<<dim3(BATCH / ROWS_PER_BLOCK), THREADS, SMEM_BYTES, stream>>>(
      feature, label, proxy, partial);
  nca_reduce<<<1, 256, 0, stream>>>(partial, out);
}
